// Net_40106404610839
// MI455X (gfx1250) — matrix operand order repaired and checked by reading, not compiled, not run on hardware
//
#include <hip/hip_runtime.h>
#include <hip/hip_bf16.h>

// ---------------------------------------------------------------------------
// CDNA5 (gfx1250) windowed-transformer forward.
// M=16 everywhere -> v_wmma_f32_16x16x32_f16.
// * Weights pre-packed f32->f16 in WMMA B-fragment order (global, L2 resident)
// * Activations mirrored in LDS as f16 with 32-aligned zero-padded strides so
//   A fragments are contiguous ds_load_b128s: no cvt, no masks in inner loops.
// * A fragments preloaded to registers when NK <= 8 (reused across N tiles).
// ---------------------------------------------------------------------------

typedef __attribute__((ext_vector_type(16))) _Float16 v16h;
typedef __attribute__((ext_vector_type(8)))  _Float16 v8h;
typedef __attribute__((ext_vector_type(4)))  _Float16 v4h;
typedef __attribute__((ext_vector_type(8)))  float    v8f;

__device__ __forceinline__ v8f wmma_f16(v16h a, v16h b, v8f c) {
    return __builtin_amdgcn_wmma_f32_16x16x32_f16(
        false, a, false, b, (short)0, c, false, false);
}

// 16 contiguous halves starting at p (8-byte aligned) -> v16h
__device__ __forceinline__ v16h load16h(const _Float16* p) {
    v16h r;
#pragma unroll
    for (int c = 0; c < 4; ++c) {
        v4h x = *reinterpret_cast<const v4h*>(p + 4 * c);
#pragma unroll
        for (int u = 0; u < 4; ++u) r[4 * c + u] = x[u];
    }
    return r;
}

// ---- A fragment (16x32) from f16 LDS act[16][ld], ld % 8 == 0 --------------
// lane (m,half): elements 0..15 = act[m][k0+16h .. +15].
__device__ __forceinline__ v16h frag_a16(const _Float16* act, int ld, int k0,
                                         int lane) {
    const int m = lane & 15, half = lane >> 4;
    const _Float16* p = act + m * ld + k0 + 16 * half;
    v16h r;
#pragma unroll
    for (int c = 0; c < 2; ++c) {
        v4h x = *reinterpret_cast<const v4h*>(p + 4 * c);
        v4h y = *reinterpret_cast<const v4h*>(p + 8 + 4 * c);
#pragma unroll
        for (int u = 0; u < 4; ++u) { r[4 * c + u] = x[u]; r[8 + 4 * c + u] = y[u]; }
    }
    return r;
}

// ---- B fragment (32x16) for Q*K^T: B(k,n) = kmat16[n][coff + k] ------------
// Contiguous run k = 16*half + 0..15; mask k >= DH (head width).
template <int DH>
__device__ __forceinline__ v16h frag_b_kT(const _Float16* buf, int ld, int coff,
                                          int lane) {
    const int n = lane & 15, half = lane >> 4;
    v16h r = load16h(buf + n * ld + coff + 16 * half);
#pragma unroll
    for (int i = 0; i < 16; ++i) {
        const int k = 16 * half + i;
        r[i] = (k < DH) ? r[i] : (_Float16)0.0f;
    }
    return r;
}

// ---- B fragment for probs*V: B(k,n) = vT[(hdh+n)*32 + k], zero-padded ------
__device__ __forceinline__ v16h frag_b_v(const _Float16* vT, int hdh, int lane) {
    const int n = lane & 15, half = lane >> 4;
    return load16h(vT + (hdh + n) * 32 + 16 * half);
}

// ---- dst = act16[16][K] @ FW + bias. EP: 0=f32 out, 1=f16 out, 2=f16+relu,
//      3=qkv (f16 out + transposed V copy for columns n >= 2*N/3) ------------
template <int K, int N, int EP, int NWAVES>
__device__ __forceinline__ void gemm_fw(float* dst32, int ld32,
                                        _Float16* dst16, int ld16, _Float16* vT,
                                        const _Float16* __restrict__ act16, int lda,
                                        const _Float16* __restrict__ FW,
                                        const float* __restrict__ bias,
                                        int wave, int lane) {
    constexpr int NT = (N + 15) / 16;
    constexpr int NK = (K + 31) / 32;
    constexpr bool GN = (N % 16) != 0;
    constexpr bool PRELOAD = (NK <= 8);
    constexpr int VOFF = (EP == 3) ? 2 * (N / 3) : 0;

    v16h areg[PRELOAD ? NK : 1];
    if constexpr (PRELOAD) {
#pragma unroll
        for (int kb = 0; kb < NK; ++kb)
            areg[kb] = frag_a16(act16, lda, kb * 32, lane);
    }

    for (int t = wave; t < NT; t += NWAVES) {
        const int n0 = t * 16;
        const _Float16* fw = FW + ((size_t)t * NK) * 512 + lane * 16;
        __builtin_prefetch(fw, 0, 1);
        v8f c = {};
#pragma unroll 4
        for (int kb = 0; kb < NK; ++kb) {
            v16h a;
            if constexpr (PRELOAD) a = areg[kb];
            else                   a = frag_a16(act16, lda, kb * 32, lane);
            v16h b = *reinterpret_cast<const v16h*>(fw + (size_t)kb * 512);
            c = wmma_f16(a, b, c);
        }
        const int n = n0 + (lane & 15), half = lane >> 4;
        if (!GN || n < N) {
            const float bv = bias[n];
            float vv[8];
#pragma unroll
            for (int r = 0; r < 8; ++r) {
                float v = c[r] + bv;
                if (EP == 2) v = fmaxf(v, 0.0f);
                vv[r] = v;
            }
            if (EP == 0) {
#pragma unroll
                for (int r = 0; r < 8; ++r) dst32[(r + 8 * half) * ld32 + n] = vv[r];
            } else {
#pragma unroll
                for (int r = 0; r < 8; ++r)
                    dst16[(r + 8 * half) * ld16 + n] = (_Float16)vv[r];
            }
            if (EP == 3 && n0 >= VOFF) {  // transposed V: vT[chan][token]
                v8h pk;
#pragma unroll
                for (int r = 0; r < 8; ++r) pk[r] = (_Float16)vv[r];
                *reinterpret_cast<v8h*>(vT + (n - VOFF) * 32 + 8 * half) = pk;
            }
        }
    }
}

// ---- parallel post-LN: h = LN(h + add)*sc + bi; also refresh f16 mirror ----
template <int D, int NTHREADS>
__device__ __forceinline__ void layernorm_par(float* h, int ldh, _Float16* h16,
                                              int ldh16, const float* add, int lda,
                                              const float* __restrict__ sc,
                                              const float* __restrict__ bi,
                                              float* scratch, float* lnmv, int tid) {
    constexpr int G = NTHREADS / 16;
    const int g = tid >> 4, sub = tid & 15;
    for (int r0 = 0; r0 < 16; r0 += G) {
        const int r = r0 + g;
        float ps = 0.0f, pq = 0.0f;
        for (int c = sub; c < D; c += 16) {
            const float x = h[r * ldh + c] + add[r * lda + c];
            ps += x; pq += x * x;
        }
        scratch[g * 16 + sub] = ps;
        scratch[G * 16 + g * 16 + sub] = pq;
        __syncthreads();
        if (sub == 0) {
            float s = 0.0f, q = 0.0f;
            for (int u = 0; u < 16; ++u) {
                s += scratch[g * 16 + u];
                q += scratch[G * 16 + g * 16 + u];
            }
            const float mean = s / (float)D;
            const float var  = q / (float)D - mean * mean;
            lnmv[g * 2]     = mean;
            lnmv[g * 2 + 1] = rsqrtf(var + 1e-5f);
        }
        __syncthreads();
        const float mean = lnmv[g * 2], inv = lnmv[g * 2 + 1];
        for (int c = sub; c < D; c += 16) {
            const float x = h[r * ldh + c] + add[r * lda + c];
            const float y = (x - mean) * inv * sc[c] + bi[c];
            h[r * ldh + c] = y;
            h16[r * ldh16 + c] = (_Float16)y;
        }
        __syncthreads();
    }
}

// ---------------------------------------------------------------------------
// Weight packer: f32 W[L][K][N] -> f16 B-fragments, zero-padded.
// ---------------------------------------------------------------------------
template <int K, int N>
__global__ void pack_weights(const float* __restrict__ W, _Float16* __restrict__ out,
                             int layers) {
    constexpr int NT = (N + 15) / 16;
    constexpr int NK = (K + 31) / 32;
    const size_t total = (size_t)layers * NT * NK * 512;
    for (size_t e = (size_t)blockIdx.x * blockDim.x + threadIdx.x; e < total;
         e += (size_t)gridDim.x * blockDim.x) {
        const int i    = (int)(e & 15);
        const int lane = (int)((e >> 4) & 31);
        const size_t blk = e >> 9;
        const int kb = (int)(blk % NK);
        const size_t t2 = blk / NK;
        const int t = (int)(t2 % NT);
        const int l = (int)(t2 / NT);
        const int k = kb * 32 + 2 * (i >> 1) + 16 * (lane >> 4) + (i & 1);
        const int n = t * 16 + (lane & 15);
        const float v = (k < K && n < N) ? W[((size_t)l * K + k) * N + n] : 0.0f;
        out[e] = (_Float16)v;
    }
}

// ---------------------------------------------------------------------------
// One stage: window-partition -> L encoder layers -> downsample/store.
// ---------------------------------------------------------------------------
template <int D, int HEADS, int LAYERS, int NWAVES, int STAGE>
__global__ __launch_bounds__(NWAVES * 32)
void swin_stage_kernel(const float* __restrict__ src, float* __restrict__ dst,
                       const _Float16* __restrict__ Fqkv,
                       const _Float16* __restrict__ Fout,
                       const _Float16* __restrict__ Fw1,
                       const _Float16* __restrict__ Fw2,
                       const float* __restrict__ qkv_b, const float* __restrict__ out_b,
                       const float* __restrict__ ln1_s, const float* __restrict__ ln1_b,
                       const float* __restrict__ b1,    const float* __restrict__ b2,
                       const float* __restrict__ ln2_s, const float* __restrict__ ln2_b,
                       const float* __restrict__ ds_w,  const float* __restrict__ ds_b) {
    constexpr int S   = 16;
    constexpr int QKV = 3 * D;
    constexpr int FF  = 4 * D;
    constexpr int DH  = D / HEADS;
    constexpr int LH  = D + 1;
    constexpr int LB  = D + 1;
    constexpr int AH  = ((D + 31) / 32) * 32;   // f16 stride for h16/o16
    constexpr int FH  = 4 * D;                  // f16 stride for b1h (>= 3D, %32==0)
    constexpr int GB  = (STAGE == 1) ? 64 : ((STAGE == 2) ? 16 : 4);
    constexpr int SRC_RES = GB * 4;
    constexpr int SRC_C   = (STAGE == 1) ? 8 : ((STAGE == 2) ? 64 : 256);
    constexpr int NTQ = (QKV + 15) / 16, NKQ = (D + 31) / 32;
    constexpr int NTO = (D + 15) / 16;
    constexpr int NTF1 = (FF + 15) / 16;
    constexpr int NTF2 = (D + 15) / 16, NKF2 = (FF + 31) / 32;

    __shared__ float sh_h [S * LH];        // f32 residual stream
    __shared__ float sh_b2[S * LB];        // f32 proj / ffn2 output (LN add)
    __shared__ float sh_sc[NWAVES * 256];  // score tiles / LN scratch
    __shared__ float sh_ln[32];
    __shared__ _Float16 sh_h16[(S + 1) * AH];   // f16 mirror of h
    __shared__ _Float16 sh_b1h[(S + 1) * FH];   // qkv (16x3D) / ffn hidden (16x4D)
    __shared__ _Float16 sh_o16[(S + 1) * AH];   // attention output
    __shared__ _Float16 sh_vT [(D + 16) * 32];  // V transposed [chan][token]
    __shared__ _Float16 sh_p16[NWAVES * 512];   // probs f16 [16][32] per wave

    const int tid  = threadIdx.x;
    const int lane = tid & 31;
    const int wave = tid >> 5;
    constexpr int NTH = NWAVES * 32;

    const int bid = blockIdx.x;
    const int wb  = bid % GB;
    const int hb  = (bid / GB) % GB;
    const int b   = bid / (GB * GB);

    // ---- zero f16 buffers once (padding stays zero forever) ----
    for (int i = tid; i < (S + 1) * AH; i += NTH) { sh_h16[i] = (_Float16)0.0f; sh_o16[i] = (_Float16)0.0f; }
    for (int i = tid; i < (S + 1) * FH; i += NTH) sh_b1h[i] = (_Float16)0.0f;
    for (int i = tid; i < (D + 16) * 32; i += NTH) sh_vT[i] = (_Float16)0.0f;
    for (int i = tid; i < NWAVES * 512; i += NTH) sh_p16[i] = (_Float16)0.0f;
    __syncthreads();

    // ---- window partition gather (jax reshape/transpose semantics) ----
    for (int idx = tid; idx < S * D; idx += NTH) {
        const int s = idx / D, ch = idx % D;
        const int flat = s * D + ch;
        const int cidx = flat >> 4;
        const int rem  = flat & 15;
        const int ii   = rem >> 2, jj = rem & 3;
        const size_t g = (((size_t)b * SRC_RES + (hb * 4 + ii)) * SRC_RES +
                          (wb * 4 + jj)) * SRC_C + cidx;
        const float v = src[g];
        sh_h[s * LH + ch]  = v;
        sh_h16[s * AH + ch] = (_Float16)v;
    }
    __syncthreads();

    const float scale = rsqrtf((float)DH);

    for (int l = 0; l < LAYERS; ++l) {
        const _Float16* fq  = Fqkv + (size_t)l * NTQ  * NKQ  * 512;
        const _Float16* fo  = Fout + (size_t)l * NTO  * NKQ  * 512;
        const _Float16* ff1 = Fw1  + (size_t)l * NTF1 * NKQ  * 512;
        const _Float16* ff2 = Fw2  + (size_t)l * NTF2 * NKF2 * 512;
        const float* Bq = qkv_b + (size_t)l * QKV;
        const float* Bo = out_b + (size_t)l * D;
        const float* S1 = ln1_s + (size_t)l * D;
        const float* G1 = ln1_b + (size_t)l * D;
        const float* B1 = b1 + (size_t)l * FF;
        const float* B2 = b2 + (size_t)l * D;
        const float* S2 = ln2_s + (size_t)l * D;
        const float* G2 = ln2_b + (size_t)l * D;

        // qkv = h @ Wqkv + b -> sh_b1h (f16) + sh_vT (transposed V)
        gemm_fw<D, QKV, 3, NWAVES>(nullptr, 0, sh_b1h, FH, sh_vT, sh_h16, AH,
                                   fq, Bq, wave, lane);
        __syncthreads();

        // ---- attention (HEADS % NWAVES == 0) ----
        float* scw = sh_sc + wave * 256;
        _Float16* p16 = sh_p16 + wave * 512;
        for (int hid = wave; hid < HEADS; hid += NWAVES) {
            v8f c = {};
            v16h qa = frag_a16(sh_b1h + hid * DH, FH, 0, lane);
            v16h kb = frag_b_kT<DH>(sh_b1h, FH, D + hid * DH, lane);
            c = wmma_f16(qa, kb, c);
            {
                const int n = lane & 15, half = lane >> 4;
#pragma unroll
                for (int r = 0; r < 8; ++r)
                    scw[(r + 8 * half) * 16 + n] = c[r] * scale;
            }
            if (lane < 16) {  // softmax over key dim; same-wave LDS in order
                float* rowp = scw + lane * 16;
                _Float16* rowp16 = p16 + lane * 32;
                float mx = rowp[0];
                for (int t2 = 1; t2 < 16; ++t2) mx = fmaxf(mx, rowp[t2]);
                float sum = 0.0f;
                for (int t2 = 0; t2 < 16; ++t2) {
                    const float e = __expf(rowp[t2] - mx);
                    rowp[t2] = e; sum += e;
                }
                const float inv = 1.0f / sum;
                for (int t2 = 0; t2 < 16; ++t2)
                    rowp16[t2] = (_Float16)(rowp[t2] * inv);
            }
            v8f c2 = {};
            v16h pa = frag_a16(p16, 32, 0, lane);
            v16h vb = frag_b_v(sh_vT, hid * DH, lane);
            c2 = wmma_f16(pa, vb, c2);
            {
                const int n = lane & 15, half = lane >> 4;
                if (n < DH) {
#pragma unroll
                    for (int r = 0; r < 8; ++r)
                        sh_o16[(r + 8 * half) * AH + hid * DH + n] = (_Float16)c2[r];
                }
            }
        }
        __syncthreads();

        gemm_fw<D, D, 0, NWAVES>(sh_b2, LB, nullptr, 0, nullptr, sh_o16, AH,
                                 fo, Bo, wave, lane);
        __syncthreads();
        layernorm_par<D, NTH>(sh_h, LH, sh_h16, AH, sh_b2, LB, S1, G1, sh_sc,
                              sh_ln, tid);

        gemm_fw<D, FF, 2, NWAVES>(nullptr, 0, sh_b1h, FH, nullptr, sh_h16, AH,
                                  ff1, B1, wave, lane);
        __syncthreads();
        gemm_fw<FF, D, 0, NWAVES>(sh_b2, LB, nullptr, 0, nullptr, sh_b1h, FH,
                                  ff2, B2, wave, lane);
        __syncthreads();
        layernorm_par<D, NTH>(sh_h, LH, sh_h16, AH, sh_b2, LB, S2, G2, sh_sc,
                              sh_ln, tid);
    }

    // ---- stage output ----
    if (STAGE == 3) {
        for (int idx = tid; idx < S * D; idx += NTH) {
            const int s = idx / D, ch = idx % D;
            dst[(((size_t)b * 4 + hb) * 4 + wb) * (16 * 256) + s * 256 + ch] =
                sh_h[s * LH + ch];
        }
    } else {
        constexpr int DD = (STAGE == 1) ? 4 : 16;
        constexpr int OC = 16 * DD;
        for (int idx = tid; idx < S * DD; idx += NTH) {
            const int s = idx / DD, c = idx % DD;
            float acc = ds_b[c];
            for (int k = 0; k < D; ++k) acc += sh_h[s * LH + k] * ds_w[k * DD + c];
            dst[(((size_t)b * GB + hb) * GB + wb) * OC + s * DD + c] = acc;
        }
    }
}

// ---------------------------------------------------------------------------
// Host launcher. d_in order (setup_inputs dict, leaves in insertion order):
//   [0] img [1] text_features
//   [2..13] params1: qkv_w qkv_b out_w out_b ln1_s ln1_b w1 b1 w2 b2 ln2_s ln2_b
//   [14..25] params2   [26..37] params3
//   [38] ds1_w [39] ds1_b [40] ds2_w [41] ds2_b
// ---------------------------------------------------------------------------
extern "C" void kernel_launch(void* const* d_in, const int* in_sizes, int n_in,
                              void* d_out, int out_size, void* d_ws, size_t ws_size,
                              hipStream_t stream) {
    (void)in_sizes; (void)n_in; (void)out_size; (void)ws_size;
    const float* img = (const float*)d_in[0];
    const float* P1[12]; const float* P2[12]; const float* P3[12];
    for (int i = 0; i < 12; ++i) {
        P1[i] = (const float*)d_in[2 + i];
        P2[i] = (const float*)d_in[14 + i];
        P3[i] = (const float*)d_in[26 + i];
    }
    const float* ds1_w = (const float*)d_in[38];
    const float* ds1_b = (const float*)d_in[39];
    const float* ds2_w = (const float*)d_in[40];
    const float* ds2_b = (const float*)d_in[41];

    float* out = (float*)d_out;
    float* ws1 = (float*)d_ws;                        // [16,64,64,64]  16 MiB
    float* ws2 = ws1 + (size_t)16 * 64 * 64 * 64;     // [16,16,16,256]  4 MiB
    _Float16* f16b = (_Float16*)(ws2 + (size_t)16 * 16 * 16 * 256);

    auto nt = [](int n) { return (n + 15) / 16; };
    auto nk = [](int k) { return (k + 31) / 32; };
    size_t off = 0;
    auto alloc = [&](int L, int K, int N) {
        _Float16* p = f16b + off;
        off += (size_t)L * nt(N) * nk(K) * 512;
        return p;
    };
    _Float16* F1q = alloc(6, 8, 24);    _Float16* F1o = alloc(6, 8, 8);
    _Float16* F1a = alloc(6, 8, 32);    _Float16* F1b = alloc(6, 32, 8);
    _Float16* F2q = alloc(6, 64, 192);  _Float16* F2o = alloc(6, 64, 64);
    _Float16* F2a = alloc(6, 64, 256);  _Float16* F2b = alloc(6, 256, 64);
    _Float16* F3q = alloc(15, 256, 768);  _Float16* F3o = alloc(15, 256, 256);
    _Float16* F3a = alloc(15, 256, 1024); _Float16* F3b = alloc(15, 1024, 256);

    auto blocks = [&](int L, int K, int N) {
        size_t tot = (size_t)L * nt(N) * nk(K) * 512;
        return (unsigned)((tot + 255) / 256);
    };
    pack_weights<8, 24><<<blocks(6, 8, 24), 256, 0, stream>>>(P1[0], F1q, 6);
    pack_weights<8, 8><<<blocks(6, 8, 8), 256, 0, stream>>>(P1[2], F1o, 6);
    pack_weights<8, 32><<<blocks(6, 8, 32), 256, 0, stream>>>(P1[6], F1a, 6);
    pack_weights<32, 8><<<blocks(6, 32, 8), 256, 0, stream>>>(P1[8], F1b, 6);
    pack_weights<64, 192><<<blocks(6, 64, 192), 256, 0, stream>>>(P2[0], F2q, 6);
    pack_weights<64, 64><<<blocks(6, 64, 64), 256, 0, stream>>>(P2[2], F2o, 6);
    pack_weights<64, 256><<<blocks(6, 64, 256), 256, 0, stream>>>(P2[6], F2a, 6);
    pack_weights<256, 64><<<blocks(6, 256, 64), 256, 0, stream>>>(P2[8], F2b, 6);
    pack_weights<256, 768><<<blocks(15, 256, 768), 256, 0, stream>>>(P3[0], F3q, 15);
    pack_weights<256, 256><<<blocks(15, 256, 256), 256, 0, stream>>>(P3[2], F3o, 15);
    pack_weights<256, 1024><<<blocks(15, 256, 1024), 256, 0, stream>>>(P3[6], F3a, 15);
    pack_weights<1024, 256><<<blocks(15, 1024, 256), 256, 0, stream>>>(P3[8], F3b, 15);

    swin_stage_kernel<8, 2, 6, 2, 1><<<16 * 64 * 64, 64, 0, stream>>>(
        img, ws1, F1q, F1o, F1a, F1b, P1[1], P1[3], P1[4], P1[5], P1[7], P1[9],
        P1[10], P1[11], ds1_w, ds1_b);
    swin_stage_kernel<64, 8, 6, 4, 2><<<16 * 16 * 16, 128, 0, stream>>>(
        ws1, ws2, F2q, F2o, F2a, F2b, P2[1], P2[3], P2[4], P2[5], P2[7], P2[9],
        P2[10], P2[11], ds2_w, ds2_b);
    swin_stage_kernel<256, 16, 15, 8, 3><<<16 * 4 * 4, 256, 0, stream>>>(
        ws2, out, F3q, F3o, F3a, F3b, P3[1], P3[3], P3[4], P3[5], P3[7], P3[9],
        P3[10], P3[11], nullptr, nullptr);
}
